// simple_rnn_37039797960789
// MI455X (gfx1250) — matrix operand order repaired and checked by reading, not compiled, not run on hardware
//
#include <hip/hip_runtime.h>
#include <hip/hip_bf16.h>

#define TSEQ 8192
#define HDIM 2048
#define GBLK 64                      // persistent blocks for the scan
#define ROWS_PER_BLK (HDIM / GBLK)   // 32 rows of Whh per block
#define PARTS 8                      // lanes cooperating on one row
#define KCHUNK (HDIM / PARTS)        // 256 MACs per lane per step

typedef __attribute__((ext_vector_type(16))) __bf16        v16bf;
typedef __attribute__((ext_vector_type(8)))  float         v8f;
typedef __attribute__((ext_vector_type(4)))  float         f32x4;
typedef __attribute__((ext_vector_type(4)))  unsigned int  u32x4;

union BfFrag { v16bf v; u32x4 q[2]; };

// ---------------------------------------------------------------------------
// Transpose + convert Wih1 (H x H fp32, row-major) -> WihT (K x N bf16),
// WihT[k][j] = Wih1[j][k].  LDS-tiled for coalesced loads and stores.
// ---------------------------------------------------------------------------
__global__ __launch_bounds__(256) void transpose_cvt_bf16(
    const float* __restrict__ W, __hip_bfloat16* __restrict__ Wt, int n) {
  __shared__ float tile[16][17];
  const int tx = threadIdx.x & 15;
  const int ty = threadIdx.x >> 4;
  const int bx = blockIdx.x * 16;
  const int by = blockIdx.y * 16;
  tile[ty][tx] = W[(size_t)(by + ty) * n + bx + tx];
  __syncthreads();
  // Wt[bx+ty][by+tx] = W[by+tx][bx+ty] = tile[tx][ty]
  Wt[(size_t)(bx + ty) * n + by + tx] = __float2bfloat16(tile[tx][ty]);
}

__global__ __launch_bounds__(256) void bias_sum_kernel(
    const float* __restrict__ b0, const float* __restrict__ b1,
    float* __restrict__ o, int n) {
  int i = blockIdx.x * blockDim.x + threadIdx.x;
  if (i < n) o[i] = b0[i] + b1[i];
}

// ---------------------------------------------------------------------------
// Persistent sequential scan:  h_t = tanh(a_t + Whh @ h_{t-1})
//   layer 0: a_t[r] = x[t]*Wih0[r] + bih0[r] + bhh0[r]   (computed inline)
//   layer 1: a_t[r] = a1[t*H + r]                         (biases pre-folded)
// Grid barrier per step via a monotonic agent-scope counter.
// ---------------------------------------------------------------------------
__global__ __launch_bounds__(256) void rnn_scan(
    const float* __restrict__ Whh,      // H x H fp32 (row-major)
    const float* __restrict__ a,        // T x H  (nullptr for layer 0)
    const float* __restrict__ x,        // SEQ    (nullptr for layer 1)
    const float* __restrict__ wih0,     // H      (nullptr for layer 1)
    const float* __restrict__ bih, const float* __restrict__ bhh,
    const float* __restrict__ h0,       // H initial hidden state
    float* __restrict__ hping,          // 2*H ping-pong buffer
    __hip_bfloat16* __restrict__ hs_bf, // T x H bf16 output (nullptr ok)
    float* __restrict__ hT,             // H final hidden state (to d_out)
    unsigned int* __restrict__ cnt,     // barrier counter (zeroed before)
    int T) {
  __shared__ __attribute__((aligned(16))) float hsm[HDIM];
  const int tid  = threadIdx.x;          // 0..255
  const int row  = tid >> 3;             // 0..31
  const int part = tid & (PARTS - 1);    // 0..7
  const int grow = blockIdx.x * ROWS_PER_BLK + row;
  const float* wrow = Whh + (size_t)grow * HDIM + part * KCHUNK;
  const float  bsum = bih[grow] + bhh[grow];
  const float  wih  = wih0 ? wih0[grow] : 0.0f;

  for (int t = 0; t < T; ++t) {
    const float* hin = (t == 0) ? h0 : hping + (size_t)((t - 1) & 1) * HDIM;
    // Stage h_{t-1} into LDS (whole vector, 8 KB).
    for (int i = tid; i < HDIM / 4; i += 256)
      ((f32x4*)hsm)[i] = ((const f32x4*)hin)[i];
    __syncthreads();

    float s = 0.0f;
    const float* hp = hsm + part * KCHUNK;
    #pragma unroll 4
    for (int k = 0; k < KCHUNK; k += 8) {
      f32x4 w0 = *(const f32x4*)(wrow + k);
      f32x4 w1 = *(const f32x4*)(wrow + k + 4);
      f32x4 hv0 = *(const f32x4*)(hp + k);
      f32x4 hv1 = *(const f32x4*)(hp + k + 4);
      s = fmaf(w0[0], hv0[0], s); s = fmaf(w0[1], hv0[1], s);
      s = fmaf(w0[2], hv0[2], s); s = fmaf(w0[3], hv0[3], s);
      s = fmaf(w1[0], hv1[0], s); s = fmaf(w1[1], hv1[1], s);
      s = fmaf(w1[2], hv1[2], s); s = fmaf(w1[3], hv1[3], s);
    }
    // Reduce 8 partial sums (lanes part=0..7 of this row, same wave).
    s += __shfl_down(s, 4, 8);
    s += __shfl_down(s, 2, 8);
    s += __shfl_down(s, 1, 8);

    if (part == 0) {
      float pre = a ? a[(size_t)t * HDIM + grow] : fmaf(x[t], wih, bsum);
      float hv = tanhf(pre + s);
      hping[(size_t)(t & 1) * HDIM + grow] = hv;
      if (hs_bf) hs_bf[(size_t)t * HDIM + grow] = __float2bfloat16(hv);
      if (t == T - 1) hT[grow] = hv;
    }
    __syncthreads();
    // Device-wide release/acquire barrier (monotonic counter).
    if (tid == 0) {
      __hip_atomic_fetch_add(cnt, 1u, __ATOMIC_ACQ_REL, __HIP_MEMORY_SCOPE_AGENT);
      const unsigned target = (unsigned)(t + 1) * (unsigned)GBLK;
      while (__hip_atomic_load(cnt, __ATOMIC_ACQUIRE, __HIP_MEMORY_SCOPE_AGENT)
             < target) {
        __builtin_amdgcn_s_sleep(2);
      }
    }
    __syncthreads();
  }
}

// ---------------------------------------------------------------------------
// Layer-1 input projection, bf16 WMMA with double-buffered LDS staging:
//   C[t][j] = sum_k A[t][k] * Wt[k][j] + bsum[j]
// A = hs0 (M x K bf16 row-major), Wt = Wih1^T (K x N bf16 row-major).
// Block tile 64x128 (8 waves, each 32x32 = 2x2 v_wmma_f32_16x16x32_bf16).
// Per K-step: global->VGPR prefetch of next tiles overlaps WMMA on current
// LDS tiles; VGPR->LDS store after barrier.  All 8 waves share the staged
// A (64x32) and B (32x128) tiles instead of re-reading L2.
// ---------------------------------------------------------------------------
__global__ __launch_bounds__(256) void rnn_in_gemm(
    const __hip_bfloat16* __restrict__ A,
    const __hip_bfloat16* __restrict__ Bt,
    const float* __restrict__ bsum,
    float* __restrict__ C, int M, int N, int K) {
  __shared__ __attribute__((aligned(16))) unsigned short lds_a[2][64 * 32];
  __shared__ __attribute__((aligned(16))) unsigned short lds_b[2][32 * 128];

  const int tid  = threadIdx.x;
  const int lane = tid & 31;
  const int w    = tid >> 5;               // 0..7
  const int wm   = w & 1;                  // 2 waves along M
  const int wn   = w >> 1;                 // 4 waves along N
  const int m0   = blockIdx.y * 64;
  const int n0   = blockIdx.x * 128;
  const int lh   = lane >> 4;              // lane half
  const int lm   = lane & 15;

  // Cooperative tile-load assignment (one b128 for A, two for B per thread).
  const int ar = tid >> 2;                 // 0..63  A tile row
  const int ak = (tid & 3) * 8;            // A k-chunk (8 bf16)
  const int bk = tid >> 3;                 // 0..31  B tile k-row
  const int bn = (tid & 7) * 16;           // B n-chunk (16 bf16)

  const __hip_bfloat16* agp = A + (size_t)(m0 + ar) * K + ak;
  const __hip_bfloat16* bgp = Bt + (size_t)bk * N + n0 + bn;

  u32x4 ra, rb0, rb1;
  ra  = *(const u32x4*)(agp);
  rb0 = *(const u32x4*)(bgp);
  rb1 = *(const u32x4*)(bgp + 8);
  *(u32x4*)(&lds_a[0][ar * 32 + ak])      = ra;
  *(u32x4*)(&lds_b[0][bk * 128 + bn])     = rb0;
  *(u32x4*)(&lds_b[0][bk * 128 + bn + 8]) = rb1;
  __syncthreads();

  v8f c00 = {}, c01 = {}, c10 = {}, c11 = {};

  for (int k0 = 0; k0 < K; k0 += 32) {
    const int  cur  = (k0 >> 5) & 1;
    const bool more = (k0 + 32) < K;
    if (more) {
      // Prefetch next K-step tiles into registers (overlaps the WMMAs).
      const __hip_bfloat16* ag = agp + (k0 + 32);
      const __hip_bfloat16* bg = bgp + (size_t)(k0 + 32) * N;
      __builtin_prefetch(ag + 32, 0, 1);
      __builtin_prefetch(bg + (size_t)32 * N, 0, 1);
      ra  = *(const u32x4*)(ag);
      rb0 = *(const u32x4*)(bg);
      rb1 = *(const u32x4*)(bg + 8);
    }

    // Build fragments from the staged LDS tiles.
    const unsigned short* la = lds_a[cur];
    const unsigned short* lb = lds_b[cur];
    BfFrag a0, a1, b0, b1;
    const int lr0 = wm * 32 + lm;          // local A row for m-tile 0
    // A fragment: lane holds row m, K = {lh*8..+7} U {16+lh*8..+7}
    a0.q[0] = *(const u32x4*)(la + lr0 * 32 + lh * 8);
    a0.q[1] = *(const u32x4*)(la + lr0 * 32 + 16 + lh * 8);
    a1.q[0] = *(const u32x4*)(la + (lr0 + 16) * 32 + lh * 8);
    a1.q[1] = *(const u32x4*)(la + (lr0 + 16) * 32 + 16 + lh * 8);
    // B fragment: lane holds column n, K = {lh*8..+7} U {16+lh*8..+7}
    const unsigned short* lbl = lb + wn * 32 + lm;
    for (int s = 0; s < 16; ++s) {
      const int kk = (s & 7) + 8 * lh + ((s >> 3) << 4);
      ((unsigned short*)&b0)[s] = lbl[kk * 128];
      ((unsigned short*)&b1)[s] = lbl[kk * 128 + 16];
    }

    c00 = __builtin_amdgcn_wmma_f32_16x16x32_bf16(false, a0.v, false, b0.v,
                                                  (short)0, c00, false, false);
    c01 = __builtin_amdgcn_wmma_f32_16x16x32_bf16(false, a0.v, false, b1.v,
                                                  (short)0, c01, false, false);
    c10 = __builtin_amdgcn_wmma_f32_16x16x32_bf16(false, a1.v, false, b0.v,
                                                  (short)0, c10, false, false);
    c11 = __builtin_amdgcn_wmma_f32_16x16x32_bf16(false, a1.v, false, b1.v,
                                                  (short)0, c11, false, false);

    __syncthreads();                       // everyone done reading buffers
    if (more) {
      const int nxt = cur ^ 1;
      *(u32x4*)(&lds_a[nxt][ar * 32 + ak])      = ra;
      *(u32x4*)(&lds_b[nxt][bk * 128 + bn])     = rb0;
      *(u32x4*)(&lds_b[nxt][bk * 128 + bn + 8]) = rb1;
    }
    __syncthreads();                       // next tiles visible
  }

  // C/D layout: VGPR r -> M = r + 8*(lane>>4); N = lane&15.
  const int m_w = m0 + wm * 32;
  const int n_w = n0 + wn * 32;
  const int gn0 = n_w + lm;
  const int gn1 = gn0 + 16;
  const float bs0 = bsum[gn0];
  const float bs1 = bsum[gn1];
  #pragma unroll
  for (int r = 0; r < 8; ++r) {
    const int gm0 = m_w + r + 8 * lh;
    const int gm1 = gm0 + 16;
    C[(size_t)gm0 * N + gn0] = c00[r] + bs0;
    C[(size_t)gm0 * N + gn1] = c01[r] + bs1;
    C[(size_t)gm1 * N + gn0] = c10[r] + bs0;
    C[(size_t)gm1 * N + gn1] = c11[r] + bs1;
  }
}

// ---------------------------------------------------------------------------
extern "C" void kernel_launch(void* const* d_in, const int* in_sizes, int n_in,
                              void* d_out, int out_size, void* d_ws,
                              size_t ws_size, hipStream_t stream) {
  const float* x     = (const float*)d_in[0];  // (1, SEQ)
  const float* hinit = (const float*)d_in[1];  // (2, 1, H)
  const float* Wih0  = (const float*)d_in[2];  // (H, 1)
  const float* Whh0  = (const float*)d_in[3];  // (H, H)
  const float* bih0  = (const float*)d_in[4];
  const float* bhh0  = (const float*)d_in[5];
  const float* Wih1  = (const float*)d_in[6];  // (H, H)
  const float* Whh1  = (const float*)d_in[7];  // (H, H)
  const float* bih1  = (const float*)d_in[8];
  const float* bhh1  = (const float*)d_in[9];
  float* out = (float*)d_out;                  // (2, 1, H)

  char* ws = (char*)d_ws;
  size_t off = 0;
  auto alloc = [&](size_t bytes) -> void* {
    off = (off + 255) & ~(size_t)255;
    void* p = ws + off;
    off += bytes;
    return p;
  };
  unsigned int*   cnt   = (unsigned int*)alloc(256);                 // 2 counters
  float*          hping = (float*)alloc((size_t)2 * HDIM * 4);
  __hip_bfloat16* hs0   = (__hip_bfloat16*)alloc((size_t)TSEQ * HDIM * 2);
  __hip_bfloat16* WihT  = (__hip_bfloat16*)alloc((size_t)HDIM * HDIM * 2);
  float*          bsum1 = (float*)alloc((size_t)HDIM * 4);
  float*          a1    = (float*)alloc((size_t)TSEQ * HDIM * 4);

  hipMemsetAsync(cnt, 0, 256, stream);

  // Prep for layer-1 GEMM (independent of layer-0 scan).
  transpose_cvt_bf16<<<dim3(HDIM / 16, HDIM / 16), 256, 0, stream>>>(
      Wih1, WihT, HDIM);
  bias_sum_kernel<<<HDIM / 256, 256, 0, stream>>>(bih1, bhh1, bsum1, HDIM);

  // Layer 0 scan: inline input projection from scalar x[t].
  rnn_scan<<<GBLK, 256, 0, stream>>>(Whh0, nullptr, x, Wih0, bih0, bhh0,
                                     hinit, hping, hs0, out, cnt + 0, TSEQ);

  // Layer 1 input projection: hs0 @ Wih1^T + (bih1 + bhh1), bf16 WMMA.
  rnn_in_gemm<<<dim3(HDIM / 128, TSEQ / 64), 256, 0, stream>>>(
      hs0, WihT, bsum1, a1, TSEQ, HDIM, HDIM);

  // Layer 1 scan.
  rnn_scan<<<GBLK, 256, 0, stream>>>(Whh1, a1, nullptr, nullptr, bih1, bhh1,
                                     hinit + HDIM, hping, nullptr, out + HDIM,
                                     cnt + 1, TSEQ);
}
